// CurvatureReg_9972914061342
// MI455X (gfx1250) — compile-verified, matrix operand order checked by reading, not run on hardware
//
#include <hip/hip_runtime.h>
#include <hip/hip_bf16.h>
#include <math.h>

typedef __attribute__((ext_vector_type(16))) _Float16 v16h;
typedef __attribute__((ext_vector_type(8)))  float    v8f;

#define S    160
#define SP   25600        // S*S
#define VOL  4096000      // S*S*S
#define NTOT 8192000      // 2*VOL
#define THRES 0.1f
#define EPSF  1e-6f

// ---------------------------------------------------------------------------
// Kernel 0: recover the separable 1-D gaussian factor from the 3-D kernel.
// gk is normalized (sum==1) and separable, so g1[t] = sum_{b,c} gk[t,b,c].
// ---------------------------------------------------------------------------
__global__ void k_prep_g1(const float* __restrict__ gk, float* __restrict__ g1) {
    int t = threadIdx.x;
    if (t < 9) {
        float s = 0.0f;
        for (int i = 0; i < 81; ++i) s += gk[t * 81 + i];
        g1[t] = s;
    }
}

// ---------------------------------------------------------------------------
// Kernel 1: conv along L (contiguous axis) via WMMA.
// One wave computes a 16(w-rows) x 16(l-cols) tile: D = A x B,
// A = 16x24 input slab (zero-padded), B[k,j] = g1[k-j] banded (K>=24 rows 0).
// Error-compensated f16 split: D = al*bh + ah*bl + ah*bh  (~f32 accuracy).
// ---------------------------------------------------------------------------
__global__ void k_convL_wmma(const float* __restrict__ in,
                             const float* __restrict__ g1,
                             float* __restrict__ out) {
    int wave = blockIdx.x * (blockDim.x >> 5) + (threadIdx.x >> 5); // < 32000
    int lane = threadIdx.x & 31;

    int lt = wave % 10;            // l tile
    int rg = wave / 10;
    int wg = rg % 10;              // w group
    int h  = (rg / 10) % S;
    int n  = rg / 1600;
    int l0 = lt * 16;
    int w0 = wg * 16;

    int mrow = lane & 15;          // A row / D column index
    int hi   = lane >> 4;          // lane group 0/1

    float g1f[9];
#pragma unroll
    for (int t = 0; t < 9; ++t) g1f[t] = g1[t];

    const float* rowp = in + (size_t)n * VOL + (size_t)h * SP + (size_t)(w0 + mrow) * S;

    v16h ah, al, bh, bl;
#pragma unroll
    for (int e = 0; e < 16; ++e) {
        // 16-bit A layout: lanes<16 hold K 0-7 & 16-23; lanes>=16 hold K 8-15 & 24-31
        int k = (hi ? 8 : 0) + (e < 8 ? e : e + 8);
        int l = l0 - 4 + k;
        float v = (k < 24 && l >= 0 && l < S) ? rowp[l] : 0.0f; // zero-pad, never garbage
        _Float16 vh = (_Float16)v;
        ah[e] = vh;
        al[e] = (_Float16)(v - (float)vh);

        // B (32x16) mirrored layout: this lane owns column n = mrow, same K map
        int wdx = k - mrow;
        float wv = (wdx >= 0 && wdx < 9) ? g1f[wdx] : 0.0f;
        _Float16 wh = (_Float16)wv;
        bh[e] = wh;
        bl[e] = (_Float16)(wv - (float)wh);
    }

    v8f c = {};
    c = __builtin_amdgcn_wmma_f32_16x16x32_f16(false, al, false, bh, (short)0, c, false, false);
    c = __builtin_amdgcn_wmma_f32_16x16x32_f16(false, ah, false, bl, (short)0, c, false, false);
    c = __builtin_amdgcn_wmma_f32_16x16x32_f16(false, ah, false, bh, (short)0, c, false, false);

    float* obase = out + (size_t)n * VOL + (size_t)h * SP;
#pragma unroll
    for (int r = 0; r < 8; ++r) {
        int m = r + hi * 8;                      // D row (w offset)
        obase[(size_t)(w0 + m) * S + (l0 + mrow)] = c[r];
    }
}

// ---------------------------------------------------------------------------
// Kernel 2/3: 1-D conv along a strided axis (W: stride=S, H: stride=SP).
// ---------------------------------------------------------------------------
template <int STRIDE>
__global__ void k_conv1d(const float* __restrict__ in,
                         const float* __restrict__ g1,
                         float* __restrict__ out) {
    int idx = blockIdx.x * 256 + threadIdx.x;     // grid sized exactly to NTOT
    int i = (idx / STRIDE) % S;
    float g[9];
#pragma unroll
    for (int t = 0; t < 9; ++t) g[t] = g1[t];
    float s = 0.0f;
#pragma unroll
    for (int t = 0; t < 9; ++t) {
        int j = i + t - 4;
        if (j >= 0 && j < S) s += g[t] * in[idx + (t - 4) * STRIDE];
    }
    out[idx] = s;
}

// jnp.gradient along one axis: central in interior, one-sided at edges.
__device__ __forceinline__ float gradf(const float* __restrict__ f, int idx, int i, int stride) {
    if (i == 0)     return f[idx + stride] - f[idx];
    if (i == S - 1) return f[idx] - f[idx - stride];
    return 0.5f * (f[idx + stride] - f[idx - stride]);
}

// ---------------------------------------------------------------------------
// Kernel 4: gm = sqrt(gx^2+gy^2+gz^2) + eps
// ---------------------------------------------------------------------------
__global__ void k_gm(const float* __restrict__ phi, float* __restrict__ gm) {
    int idx = blockIdx.x * 256 + threadIdx.x;
    int l = idx % S;
    int w = (idx / S) % S;
    int h = (idx / SP) % S;
    float gx = gradf(phi, idx, h, SP);
    float gy = gradf(phi, idx, w, S);
    float gz = gradf(phi, idx, l, 1);
    gm[idx] = sqrtf(gx * gx + gy * gy + gz * gz) + EPSF;
}

// ---------------------------------------------------------------------------
// Kernel 5: k = sum_axis grad_axis( grad_axis(phi)/gm )
// ---------------------------------------------------------------------------
__global__ void k_curv(const float* __restrict__ phi, const float* __restrict__ gm,
                       float* __restrict__ kout) {
    int idx = blockIdx.x * 256 + threadIdx.x;
    int l = idx % S;
    int w = (idx / S) % S;
    int h = (idx / SP) % S;

    float kv = 0.0f;
    {   // axis H
        auto nf = [&](int ii) { int id2 = idx + (ii - h) * SP; return gradf(phi, id2, ii, SP) / gm[id2]; };
        kv += (h == 0) ? (nf(1) - nf(0)) : (h == S - 1) ? (nf(S - 1) - nf(S - 2))
                       : 0.5f * (nf(h + 1) - nf(h - 1));
    }
    {   // axis W
        auto nf = [&](int ii) { int id2 = idx + (ii - w) * S; return gradf(phi, id2, ii, S) / gm[id2]; };
        kv += (w == 0) ? (nf(1) - nf(0)) : (w == S - 1) ? (nf(S - 1) - nf(S - 2))
                       : 0.5f * (nf(w + 1) - nf(w - 1));
    }
    {   // axis L
        auto nf = [&](int ii) { int id2 = idx + (ii - l); return gradf(phi, id2, ii, 1) / gm[id2]; };
        kv += (l == 0) ? (nf(1) - nf(0)) : (l == S - 1) ? (nf(S - 1) - nf(S - 2))
                       : 0.5f * (nf(l + 1) - nf(l - 1));
    }
    kout[idx] = kv;
}

// ---------------------------------------------------------------------------
// Kernel 6: score = min(|d_L|,|d_W|,|d_H|) (zero-padded central diffs of k),
// mask = interior-restricted 3x3x3 dilation of (pred>=0.1); block-sum partials.
// ---------------------------------------------------------------------------
__global__ void k_score(const float* __restrict__ kf, const float* __restrict__ pred,
                        float* __restrict__ partial) {
    int idx = blockIdx.x * 256 + threadIdx.x;
    int l = idx % S;
    int w = (idx / S) % S;
    int h = (idx / SP) % S;

    float d1 = ((l + 1 < S) ? kf[idx + 1]  : 0.0f) - ((l >= 1) ? kf[idx - 1]  : 0.0f);
    float d2 = ((w + 1 < S) ? kf[idx + S]  : 0.0f) - ((w >= 1) ? kf[idx - S]  : 0.0f);
    float d3 = ((h + 1 < S) ? kf[idx + SP] : 0.0f) - ((h >= 1) ? kf[idx - SP] : 0.0f);
    float sc = fminf(fminf(fabsf(d1), fabsf(d2)), fabsf(d3));

    float m = 0.0f;
    if (h >= 5 && h <= S - 6 && w >= 5 && w <= S - 6 && l >= 5 && l <= S - 6) {
        // interior => all 27 neighbors are in-bounds
#pragma unroll
        for (int dh = -1; dh <= 1; ++dh)
#pragma unroll
            for (int dw = -1; dw <= 1; ++dw)
#pragma unroll
                for (int dl = -1; dl <= 1; ++dl)
                    if (pred[idx + dh * SP + dw * S + dl] >= THRES) m = 1.0f;
    }

    __shared__ float red[256];
    red[threadIdx.x] = sc * m;
    __syncthreads();
#pragma unroll
    for (int s2 = 128; s2 > 0; s2 >>= 1) {
        if (threadIdx.x < s2) red[threadIdx.x] += red[threadIdx.x + s2];
        __syncthreads();
    }
    if (threadIdx.x == 0) partial[blockIdx.x] = red[0];
}

// ---------------------------------------------------------------------------
// Kernel 7: deterministic final reduction of 32000 partials -> mean.
// ---------------------------------------------------------------------------
__global__ void k_final(const float* __restrict__ partial, float* __restrict__ out) {
    __shared__ float red[256];
    float s = 0.0f;
    for (int i = threadIdx.x; i < 32000; i += 256) s += partial[i];
    red[threadIdx.x] = s;
    __syncthreads();
#pragma unroll
    for (int s2 = 128; s2 > 0; s2 >>= 1) {
        if (threadIdx.x < s2) red[threadIdx.x] += red[threadIdx.x + s2];
        __syncthreads();
    }
    if (threadIdx.x == 0) out[0] = red[0] / (float)NTOT;
}

// ---------------------------------------------------------------------------
extern "C" void kernel_launch(void* const* d_in, const int* in_sizes, int n_in,
                              void* d_out, int out_size, void* d_ws, size_t ws_size,
                              hipStream_t stream) {
    (void)in_sizes; (void)n_in; (void)out_size; (void)ws_size;
    const float* pred = (const float*)d_in[0];   // (2,1,160,160,160) f32
    const float* gk   = (const float*)d_in[1];   // (1,1,9,9,9) f32

    float* ws   = (float*)d_ws;
    float* g1   = ws;                  // 9 floats (64 reserved)
    float* part = ws + 64;             // 32000 floats
    float* A    = ws + 40960;          // 160^3 * 2
    float* B    = A + NTOT;
    float* C    = B + NTOT;

    k_prep_g1<<<1, 32, 0, stream>>>(gk, g1);

    // separable gaussian: L (WMMA), then W, then H  -> phi in C
    k_convL_wmma<<<4000, 256, 0, stream>>>(pred, g1, A);
    k_conv1d<S ><<<NTOT / 256, 256, 0, stream>>>(A, g1, B);
    k_conv1d<SP><<<NTOT / 256, 256, 0, stream>>>(B, g1, C);

    k_gm   <<<NTOT / 256, 256, 0, stream>>>(C, A);        // |grad phi|+eps -> A
    k_curv <<<NTOT / 256, 256, 0, stream>>>(C, A, B);     // curvature -> B
    k_score<<<NTOT / 256, 256, 0, stream>>>(B, pred, part);
    k_final<<<1, 256, 0, stream>>>(part, (float*)d_out);
}
